// LlamaAttention_80917183856891
// MI455X (gfx1250) — matrix operand order repaired and checked by reading, not compiled, not run on hardware
//
#include <hip/hip_runtime.h>

// ---------------------------------------------------------------------------
// LlamaAttention prefill-with-cache for MI455X (gfx1250), wave32 + WMMA bf16.
// B=4, S=128, D=4096, H=32, NKV=8, HD=128, CACHE=2048, TT=2176.
// ---------------------------------------------------------------------------

#define B_     4
#define S_     128
#define D_     4096
#define H_     32
#define NKV_   8
#define HD_    128
#define CACHE_ 2048
#define TT_    (CACHE_ + S_)   // 2176

typedef __bf16 bf16;
typedef __attribute__((ext_vector_type(16))) __bf16 v16bf;
typedef __attribute__((ext_vector_type(8)))  __bf16 v8bf;
typedef __attribute__((ext_vector_type(4)))  __bf16 v4bf;
typedef __attribute__((ext_vector_type(8)))  float  v8f;

#define CAT16(lo, hi) __builtin_shufflevector((lo), (hi), \
    0,1,2,3,4,5,6,7,8,9,10,11,12,13,14,15)

static __device__ __forceinline__ v8f v8f_zero() {
  v8f z = {0.f,0.f,0.f,0.f,0.f,0.f,0.f,0.f};
  return z;
}

// D = A(16x32 bf16) * B(32x16 bf16) + C(16x16 f32)
static __device__ __forceinline__ v8f wmma_bf16(v16bf a, v16bf b, v8f c) {
  return __builtin_amdgcn_wmma_f32_16x16x32_bf16(
      /*neg_a=*/false, a, /*neg_b=*/false, b,
      /*c_mod=*/(short)0, c, /*reuse_a=*/false, /*reuse_b=*/false);
}

// ---------------------------------------------------------------------------
// Tiled GEMM: C[M,N] = A[M,K] * W[N,K]^T   (both operands K-contiguous = "NT")
// fp32 in/out, bf16 WMMA compute. Block: 256 threads = 8 waves.
// Tile: BM=128, BN=128, BK=32. Wave grid 2(M) x 4(N): each wave 64x32.
// Double-buffered LDS ping-pong: ONE barrier per K-iteration; the next tile's
// global_load_b128s are issued right after the barrier so LOADcnt covers them
// across the whole WMMA block. LDS rows padded to 40 bf16 (80B) so every
// fragment load is 16B-aligned.
// ---------------------------------------------------------------------------
__global__ __launch_bounds__(256) void gemm_nt_kernel(
    const float* __restrict__ A, const float* __restrict__ W,
    float* __restrict__ C, int M, int N, int K)
{
  __shared__ bf16 As[2][128][40];
  __shared__ bf16 Bs[2][128][40];

  const int n0 = blockIdx.x * 128;
  const int m0 = blockIdx.y * 128;
  const int tid  = threadIdx.x;
  const int wave = tid >> 5;
  const int lane = tid & 31;
  const int lr = lane & 15;      // lane % 16
  const int lh = lane >> 4;      // lane half (0/1)
  const int wm = (wave & 1) * 64;
  const int wn = (wave >> 1) * 32;

  // Each thread stages 4 float4 of A and 4 of W per 128x32 tile.
  const int srow[4] = { (tid + 0*256) >> 3, (tid + 1*256) >> 3,
                        (tid + 2*256) >> 3, (tid + 3*256) >> 3 };
  const int sc4 = (tid & 7) << 2;

  v8f acc[4][2];
#pragma unroll
  for (int fm = 0; fm < 4; ++fm)
#pragma unroll
    for (int fn = 0; fn < 2; ++fn) acc[fm][fn] = v8f_zero();

  float4 ar[4], br[4];

  // ---- prologue: stage tile 0 into buffer 0 ----------------------------
#pragma unroll
  for (int it = 0; it < 4; ++it) {
    ar[it] = *(const float4*)(A + (size_t)(m0 + srow[it]) * K + sc4);
    br[it] = *(const float4*)(W + (size_t)(n0 + srow[it]) * K + sc4);
  }
#pragma unroll
  for (int it = 0; it < 4; ++it) {
    v4bf a4 = { (bf16)ar[it].x, (bf16)ar[it].y, (bf16)ar[it].z, (bf16)ar[it].w };
    v4bf b4 = { (bf16)br[it].x, (bf16)br[it].y, (bf16)br[it].z, (bf16)br[it].w };
    *(v4bf*)&As[0][srow[it]][sc4] = a4;
    *(v4bf*)&Bs[0][srow[it]][sc4] = b4;
  }

  int buf = 0;
  for (int kk = 0; kk < K; kk += 32) {
    __syncthreads();   // buffer `buf` fully written; buf^1 reads all retired

    const bool more = (kk + 32) < K;
    if (more) {        // issue next tile's loads immediately (latency hiding)
      const int kn = kk + 32;
#pragma unroll
      for (int it = 0; it < 4; ++it) {
        ar[it] = *(const float4*)(A + (size_t)(m0 + srow[it]) * K + kn + sc4);
        br[it] = *(const float4*)(W + (size_t)(n0 + srow[it]) * K + kn + sc4);
      }
      if (kn + 32 < K) {
        __builtin_prefetch(A + (size_t)(m0 + srow[0]) * K + kn + 32 + sc4, 0, 0);
        __builtin_prefetch(W + (size_t)(n0 + srow[0]) * K + kn + 32 + sc4, 0, 0);
      }
    }

    // A-fragment (16x32): lane row = lr, elems 0..7 @ K=lh*8, 8..15 @ K=16+lh*8
    v16bf af[4];
#pragma unroll
    for (int fm = 0; fm < 4; ++fm) {
      const bf16* p = &As[buf][wm + fm * 16 + lr][0];
      v8bf lo = *(const v8bf*)(p + lh * 8);
      v8bf hi = *(const v8bf*)(p + 16 + lh * 8);
      af[fm] = CAT16(lo, hi);
    }
    // B-fragment (32x16): lane col = lr, elems 0..7 @ K=lh*8, 8..15 @ K=16+lh*8
    v16bf bfr[2];
#pragma unroll
    for (int fn = 0; fn < 2; ++fn) {
      const bf16* p = &Bs[buf][wn + fn * 16 + lr][0];
      v8bf lo = *(const v8bf*)(p + lh * 8);
      v8bf hi = *(const v8bf*)(p + 16 + lh * 8);
      bfr[fn] = CAT16(lo, hi);
    }
#pragma unroll
    for (int fm = 0; fm < 4; ++fm)
#pragma unroll
      for (int fn = 0; fn < 2; ++fn)
        acc[fm][fn] = wmma_bf16(af[fm], bfr[fn], acc[fm][fn]);

    if (more) {        // convert + store into the alternate buffer
      const int nb = buf ^ 1;
#pragma unroll
      for (int it = 0; it < 4; ++it) {
        v4bf a4 = { (bf16)ar[it].x, (bf16)ar[it].y, (bf16)ar[it].z, (bf16)ar[it].w };
        v4bf b4 = { (bf16)br[it].x, (bf16)br[it].y, (bf16)br[it].z, (bf16)br[it].w };
        *(v4bf*)&As[nb][srow[it]][sc4] = a4;
        *(v4bf*)&Bs[nb][srow[it]][sc4] = b4;
      }
      buf = nb;
    }
  }

  // C/D layout: VGPR r holds row (r + 8*lh), column = lr.
#pragma unroll
  for (int fm = 0; fm < 4; ++fm)
#pragma unroll
    for (int fn = 0; fn < 2; ++fn)
#pragma unroll
      for (int r = 0; r < 8; ++r) {
        const int row = m0 + wm + fm * 16 + r + 8 * lh;
        const int col = n0 + wn + fn * 16 + lr;
        C[(size_t)row * N + col] = acc[fm][fn][r];
      }
}

// ---------------------------------------------------------------------------
// Convert fp32 KV caches into WMMA-friendly bf16 layouts:
//   k_caches (NKV,B,HD,CACHE) -> Kall[b][kv][t][d]   (QK^T B-matrix rows)
//   v_caches (NKV,B,CACHE,HD) -> Vt  [b][kv][d][t]   (PV   B-matrix rows)
// ---------------------------------------------------------------------------
__global__ void cache_prep_kernel(const float* __restrict__ kc,
                                  const float* __restrict__ vc,
                                  bf16* __restrict__ Kall,
                                  bf16* __restrict__ Vt)
{
  const long NCK = (long)NKV_ * B_ * HD_ * CACHE_;   // 8,388,608
  long i = (long)blockIdx.x * blockDim.x + threadIdx.x;
  if (i < NCK) {
    // i = ((kv*B + b)*HD + d)*CACHE + t
    const int t = (int)(i & (CACHE_ - 1));
    long x = i >> 11;
    const int d = (int)(x & (HD_ - 1)); x >>= 7;
    const int b = (int)(x & (B_ - 1));
    const int kv = (int)(x >> 2);
    Kall[((size_t)((b * NKV_ + kv) * TT_) + t) * HD_ + d] = (bf16)kc[i];
  } else {
    const long ii = i - NCK;
    // ii = ((kv*B + b)*CACHE + t)*HD + d
    const int d = (int)(ii & (HD_ - 1));
    long x = ii >> 7;
    const int t = (int)(x & (CACHE_ - 1)); x >>= 11;
    const int b = (int)(x & (B_ - 1));
    const int kv = (int)(x >> 2);
    Vt[((size_t)((b * NKV_ + kv) * HD_) + d) * TT_ + t] = (bf16)vc[ii];
  }
}

// ---------------------------------------------------------------------------
// RoPE + repack. Reference rope: out[0:64] = xr*cos - xi*sin,
// out[64:128] = xr*sin + xi*cos with xr=x[2j], xi=x[2j+1].
// Also emits k_t (B,NKV,HD,S) and v_t (B,NKV,S,HD) fp32 outputs.
// ---------------------------------------------------------------------------
__global__ void rope_pack_kernel(
    const float* __restrict__ Qf, const float* __restrict__ Kf,
    const float* __restrict__ Vf,
    const float* __restrict__ fcos, const float* __restrict__ fsin,
    bf16* __restrict__ Qb, bf16* __restrict__ Kall, bf16* __restrict__ Vt,
    float* __restrict__ kt_out, float* __restrict__ vt_out)
{
  const int NQ = B_ * S_ * H_ * (HD_ / 2);     // 1,048,576
  const int NK = B_ * S_ * NKV_ * (HD_ / 2);   //   262,144
  const int NV = B_ * S_ * NKV_ * HD_;         //   524,288
  const int i = blockIdx.x * blockDim.x + threadIdx.x;

  if (i < NQ) {
    const int j = i & 63; int t = i >> 6;
    const int h = t & (H_ - 1); t >>= 5;
    const int s = t & (S_ - 1);
    const int b = t >> 7;
    const float* src = Qf + (size_t)(b * S_ + s) * D_ + h * HD_;
    const float xr = src[2 * j], xi = src[2 * j + 1];
    const float c = fcos[s * 64 + j], sn = fsin[s * 64 + j];
    bf16* dst = Qb + ((size_t)((b * H_ + h) * S_) + s) * HD_;
    dst[j]      = (bf16)(xr * c - xi * sn);
    dst[64 + j] = (bf16)(xr * sn + xi * c);
  } else if (i < NQ + NK) {
    const int ii = i - NQ;
    const int j = ii & 63; int t = ii >> 6;
    const int kv = t & (NKV_ - 1); t >>= 3;
    const int s = t & (S_ - 1);
    const int b = t >> 7;
    const float* src = Kf + (size_t)(b * S_ + s) * (NKV_ * HD_) + kv * HD_;
    const float xr = src[2 * j], xi = src[2 * j + 1];
    const float c = fcos[s * 64 + j], sn = fsin[s * 64 + j];
    const float r0 = xr * c - xi * sn;
    const float r1 = xr * sn + xi * c;
    kt_out[((size_t)((b * NKV_ + kv) * HD_) + j) * S_ + s]      = r0;
    kt_out[((size_t)((b * NKV_ + kv) * HD_) + 64 + j) * S_ + s] = r1;
    bf16* dst = Kall + ((size_t)((b * NKV_ + kv) * TT_) + CACHE_ + s) * HD_;
    dst[j]      = (bf16)r0;
    dst[64 + j] = (bf16)r1;
  } else if (i < NQ + NK + NV) {
    const int ii = i - NQ - NK;
    const int d = ii & (HD_ - 1); int t = ii >> 7;
    const int kv = t & (NKV_ - 1); t >>= 3;
    const int s = t & (S_ - 1);
    const int b = t >> 7;
    const float val = Vf[(size_t)(b * S_ + s) * (NKV_ * HD_) + kv * HD_ + d];
    vt_out[((size_t)((b * NKV_ + kv) * S_) + s) * HD_ + d] = val;
    Vt[((size_t)((b * NKV_ + kv) * HD_) + d) * TT_ + CACHE_ + s] = (bf16)val;
  }
}

// ---------------------------------------------------------------------------
// Flash attention, one workgroup per (b,h), 8 waves x 16 query rows.
// K/V fragments are read straight from global (L2-resident: 36 MB bf16,
// 192 MB L2). Online softmax per 32-column chunk; P is bounced through a
// per-wave LDS scratch to convert the C-matrix layout into an A-fragment.
// ---------------------------------------------------------------------------
__global__ __launch_bounds__(256) void attn_kernel(
    const bf16* __restrict__ Qb, const bf16* __restrict__ Kall,
    const bf16* __restrict__ Vt, const float* __restrict__ mask,
    float* __restrict__ Y)
{
  __shared__ bf16 Pl[8][16][40];   // per-wave 16x32 P tile, padded rows

  const int h = blockIdx.x, b = blockIdx.y;
  const int kv = h >> 2;                       // H/NKV = 4
  const int tid = threadIdx.x;
  const int wave = tid >> 5, lane = tid & 31;
  const int lr = lane & 15, lh = lane >> 4;
  const int m0 = wave * 16;                    // this wave's query-row base

  const bf16* Qp = Qb   + ((size_t)(b * H_ + h) * S_) * HD_;
  const bf16* Kp = Kall + ((size_t)(b * NKV_ + kv) * TT_) * HD_;
  const bf16* Vp = Vt   + ((size_t)(b * NKV_ + kv) * HD_) * TT_;
  const float* Mp = mask + (size_t)b * S_ * TT_;

  // Persistent Q A-fragments: 16 rows x 128 K as 4 fragments.
  v16bf qa[4];
#pragma unroll
  for (int kf = 0; kf < 4; ++kf) {
    const bf16* p = Qp + (size_t)(m0 + lr) * HD_ + kf * 32;
    v8bf lo = *(const v8bf*)(p + lh * 8);
    v8bf hi = *(const v8bf*)(p + 16 + lh * 8);
    qa[kf] = CAT16(lo, hi);
  }

  float mrow[8], lrow[8], alpha[8];
  v8f o[8];
#pragma unroll
  for (int r = 0; r < 8; ++r) { mrow[r] = -3.0e38f; lrow[r] = 0.f; }
#pragma unroll
  for (int dt = 0; dt < 8; ++dt) o[dt] = v8f_zero();

  const float scale = 0.08838834764831845f;    // 1/sqrt(HD)

  for (int t0 = 0; t0 < TT_; t0 += 32) {
    // ---- S = Q * K^T for two 16-column tiles --------------------------
    v8f s0 = v8f_zero(), s1 = v8f_zero();
#pragma unroll
    for (int kf = 0; kf < 4; ++kf) {
      const bf16* k0 = Kp + (size_t)(t0 + lr) * HD_ + kf * 32 + lh * 8;
      v8bf a0 = *(const v8bf*)k0;
      v8bf a1 = *(const v8bf*)(k0 + 16);
      s0 = wmma_bf16(qa[kf], CAT16(a0, a1), s0);
      const bf16* k1 = Kp + (size_t)(t0 + 16 + lr) * HD_ + kf * 32 + lh * 8;
      v8bf b0 = *(const v8bf*)k1;
      v8bf b1 = *(const v8bf*)(k1 + 16);
      s1 = wmma_bf16(qa[kf], CAT16(b0, b1), s1);
    }

    // ---- scale + mask + online softmax (row-wise over 16-lane groups) -
#pragma unroll
    for (int r = 0; r < 8; ++r) {
      const int srow = m0 + r + 8 * lh;
      float x0 = s0[r] * scale + Mp[(size_t)srow * TT_ + t0 + lr];
      float x1 = s1[r] * scale + Mp[(size_t)srow * TT_ + t0 + 16 + lr];
      float mx = fmaxf(x0, x1);
#pragma unroll
      for (int off = 1; off < 16; off <<= 1)
        mx = fmaxf(mx, __shfl_xor(mx, off, 32));
      const float mnew = fmaxf(mrow[r], mx);
      alpha[r] = __expf(mrow[r] - mnew);
      const float p0 = __expf(x0 - mnew);
      const float p1 = __expf(x1 - mnew);
      float ps = p0 + p1;
#pragma unroll
      for (int off = 1; off < 16; off <<= 1)
        ps += __shfl_xor(ps, off, 32);
      lrow[r] = lrow[r] * alpha[r] + ps;
      mrow[r] = mnew;
      Pl[wave][r + 8 * lh][lr]      = (bf16)p0;   // C layout -> plain tile
      Pl[wave][r + 8 * lh][16 + lr] = (bf16)p1;
    }

    // ---- reload P as an A-fragment (same-wave LDS, DScnt-ordered) -----
    const bf16* pp = &Pl[wave][lr][0];
    v8bf plo = *(const v8bf*)(pp + lh * 8);
    v8bf phi = *(const v8bf*)(pp + 16 + lh * 8);
    v16bf pa = CAT16(plo, phi);

    // ---- O = alpha*O + P * V  (8 output d-tiles) ----------------------
#pragma unroll
    for (int dt = 0; dt < 8; ++dt) {
      const bf16* vp = Vp + (size_t)(dt * 16 + lr) * TT_ + t0 + lh * 8;
      v8bf vlo = *(const v8bf*)vp;
      v8bf vhi = *(const v8bf*)(vp + 16);
      v8f oo = o[dt];
#pragma unroll
      for (int r = 0; r < 8; ++r) oo[r] *= alpha[r];
      o[dt] = wmma_bf16(pa, CAT16(vlo, vhi), oo);
    }
  }

  // ---- normalize and store Y[b][s][h*HD + d] (fp32) -------------------
#pragma unroll
  for (int dt = 0; dt < 8; ++dt)
#pragma unroll
    for (int r = 0; r < 8; ++r) {
      const int srow = m0 + r + 8 * lh;
      Y[(size_t)(b * S_ + srow) * (H_ * HD_) + h * HD_ + dt * 16 + lr] =
          o[dt][r] / lrow[r];
    }
}

// ---------------------------------------------------------------------------
// Host launcher. Inputs (setup_inputs order):
// 0 hidden_states 1 freqs_cos 2 freqs_sin 3 atten_mask 4 k_caches 5 v_caches
// 6 wq 7 wk 8 wv 9 wo.  Outputs: y | k_t | v_t concatenated flat (fp32).
// ---------------------------------------------------------------------------
extern "C" void kernel_launch(void* const* d_in, const int* in_sizes, int n_in,
                              void* d_out, int out_size, void* d_ws, size_t ws_size,
                              hipStream_t stream)
{
  (void)in_sizes; (void)n_in; (void)out_size; (void)ws_size;

  const float* hs   = (const float*)d_in[0];
  const float* fcos = (const float*)d_in[1];
  const float* fsin = (const float*)d_in[2];
  const float* mask = (const float*)d_in[3];
  const float* kc   = (const float*)d_in[4];
  const float* vc   = (const float*)d_in[5];
  const float* wq   = (const float*)d_in[6];
  const float* wk   = (const float*)d_in[7];
  const float* wv   = (const float*)d_in[8];
  const float* wo   = (const float*)d_in[9];

  float* yout   = (float*)d_out;                          // (B,S,D)
  float* kt_out = yout + (size_t)B_ * S_ * D_;            // (B,NKV,HD,S)
  float* vt_out = kt_out + (size_t)B_ * NKV_ * HD_ * S_;  // (B,NKV,S,HD)

  // Workspace layout (bytes, 256-aligned):
  char* w = (char*)d_ws;
  float* Qf  = (float*)(w + 0);          // (B*S, 4096) fp32:  8,388,608 B
  float* Kf  = (float*)(w + 8388608);    // (B*S, 1024) fp32:  2,097,152 B
  float* Vf  = (float*)(w + 10485760);   // (B*S, 1024) fp32:  2,097,152 B
  float* Yf  = (float*)(w + 12582912);   // (B*S, 4096) fp32:  8,388,608 B
  bf16*  Qb  = (bf16*) (w + 20971520);   // (B,H,S,HD)  bf16:  4,194,304 B
  bf16*  Kall= (bf16*) (w + 25165824);   // (B,NKV,TT,HD) bf16: 17,825,792 B
  bf16*  Vt  = (bf16*) (w + 42991616);   // (B,NKV,HD,TT) bf16: 17,825,792 B
  // total: 60,817,408 B

  const int M = B_ * S_;   // 512

  // 1) KV caches -> bf16 layouts (independent; 16.8M elements / 256)
  cache_prep_kernel<<<65536, 256, 0, stream>>>(kc, vc, Kall, Vt);

  // 2-4) QKV projections (NT GEMM, bf16 WMMA)
  gemm_nt_kernel<<<dim3(D_ / 128, M / 128), 256, 0, stream>>>(hs, wq, Qf, M, D_, D_);
  gemm_nt_kernel<<<dim3((NKV_ * HD_) / 128, M / 128), 256, 0, stream>>>(hs, wk, Kf, M, NKV_ * HD_, D_);
  gemm_nt_kernel<<<dim3((NKV_ * HD_) / 128, M / 128), 256, 0, stream>>>(hs, wv, Vf, M, NKV_ * HD_, D_);

  // 5) RoPE + repack + k_t/v_t outputs
  const int nrope = B_ * S_ * H_ * 64 + B_ * S_ * NKV_ * 64 + B_ * S_ * NKV_ * 128;
  rope_pack_kernel<<<(nrope + 255) / 256, 256, 0, stream>>>(
      Qf, Kf, Vf, fcos, fsin, Qb, Kall, Vt, kt_out, vt_out);

  // 6) Flash attention per (b,h)
  attn_kernel<<<dim3(H_, B_), 256, 0, stream>>>(Qb, Kall, Vt, mask, Yf);

  // 7) Output projection straight into d_out
  gemm_nt_kernel<<<dim3(D_ / 128, M / 128), 256, 0, stream>>>(Yf, wo, yout, M, D_, D_);
}
